// HGT_47614007443626
// MI455X (gfx1250) — matrix operand order repaired and checked by reading, not compiled, not run on hardware
//
#include <hip/hip_runtime.h>
#include <math.h>

// ---------------------------------------------------------------------------
// CDNA5 (gfx1250, wave32) HGT forward.
//   - Dense projections via v_wmma_f32_16x16x32_bf16 (bf16 in, f32 acc),
//     128x64 block tile, 32x64 wave tile (8 WMMA per wave per K-step).
//   - bf16 staging packed to 32-bit LDS stores.
//   - Relation matrices folded into combined 128x128 weights (kk/vv are
//     single WMMA GEMMs from h; k/v never materialized).
//   - Edge softmax in 3 memory-bound passes with native f32/i32 atomics.
// ---------------------------------------------------------------------------

typedef __attribute__((ext_vector_type(16))) __bf16 v16bf;
typedef __attribute__((ext_vector_type(8)))  float  v8f;

#define HIDC  128
#define NHEAD 8

// float -> bf16 bits, round-to-nearest-even
__device__ inline unsigned bfbits(float f) {
  unsigned u = __float_as_uint(f);
  return (u + 0x7FFFu + ((u >> 16) & 1u)) >> 16;
}
__device__ inline unsigned pk2bf(float lo, float hi) {
  return bfbits(lo) | (bfbits(hi) << 16);
}

__device__ inline float gelu_exact(float x) {
  return 0.5f * x * (1.0f + erff(x * 0.70710678118654752f));
}

// ---------------------------------------------------------------------------
// WMMA GEMM: C[M x Nf] = epi( act(A)[M x K] @ W[K x Nf] + bias )
//   128 threads = 4 waves; block tile 128x64; wave tile 32x64
//   a_act: 0 = identity, 1 = exact GELU (applied while staging A)
//   epi:   0 = +bias, 1 = +bias,ReLU, 2 = skip blend  g*(v+b)+(1-g)*hold
// ---------------------------------------------------------------------------
#define BM 128
#define BN 64
#define BK 32
#define KP 20   // row stride in uints: 40 bf16 (32 + 8 pad) = 20 dwords

__global__ __launch_bounds__(128)
void gemm_bf16_wmma(const float* __restrict__ A, const float* __restrict__ W,
                    const float* __restrict__ bias, float* __restrict__ C,
                    int M, int K, int Nf, int a_act, int epi,
                    const float* __restrict__ hold,
                    const float* __restrict__ skipptr) {
  __shared__ unsigned sA[BM][KP];   // bf16 pairs along K, padded rows
  __shared__ unsigned sB[BN][KP];   // transposed [n][k], bf16 pairs along K

  const int t    = threadIdx.x;
  const int wave = t >> 5;
  const int lane = t & 31;
  const int m0   = blockIdx.x * BM;
  const int n0   = blockIdx.y * BN;

  v8f acc[2][4] = {};

  const int aCol  = (t & 7) * 4;   // 0..28 (float4 along K)
  const int aRow0 = t >> 3;        // 0..15
  const int bPair = t >> 4;        // 0..7  (k-pair index)
  const int bN    = (t & 15) * 4;  // 0..60 (float4 along N)

  for (int k0 = 0; k0 < K; k0 += BK) {
    // ---- stage A tile 128x32 (f32 -> packed bf16, optional GELU) ----
#pragma unroll
    for (int rr = 0; rr < BM; rr += 16) {
      int r  = aRow0 + rr;
      int gm = m0 + r;
      float4 v = make_float4(0.f, 0.f, 0.f, 0.f);
      if (gm < M) v = *(const float4*)(A + (size_t)gm * K + k0 + aCol);
      if (a_act) {
        v.x = gelu_exact(v.x); v.y = gelu_exact(v.y);
        v.z = gelu_exact(v.z); v.w = gelu_exact(v.w);
      }
      sA[r][(aCol >> 1) + 0] = pk2bf(v.x, v.y);
      sA[r][(aCol >> 1) + 1] = pk2bf(v.z, v.w);
    }
    // ---- stage B tile 32x64 transposed: two k-rows -> packed k-pairs ----
#pragma unroll
    for (int kc = 0; kc < 16; kc += 8) {
      int pr = bPair + kc;                 // pair index 0..15
      int kr = k0 + pr * 2;
      float4 r0 = *(const float4*)(W + (size_t)kr * Nf + n0 + bN);
      float4 r1 = *(const float4*)(W + (size_t)(kr + 1) * Nf + n0 + bN);
      sB[bN + 0][pr] = pk2bf(r0.x, r1.x);
      sB[bN + 1][pr] = pk2bf(r0.y, r1.y);
      sB[bN + 2][pr] = pk2bf(r0.z, r1.z);
      sB[bN + 3][pr] = pk2bf(r0.w, r1.w);
    }
    __syncthreads();

    // ---- A fragments: lane L holds M=L%16, contiguous K half ----
    const int akb = (lane < 16) ? 0 : 16;
    v16bf af[2];
#pragma unroll
    for (int mt = 0; mt < 2; ++mt) {
      const __bf16* row = (const __bf16*)sA[wave * 32 + mt * 16 + (lane & 15)];
#pragma unroll
      for (int i = 0; i < 8; ++i) {
        af[mt][i]     = row[akb + i];
        af[mt][i + 8] = row[akb + 8 + i];
      }
    }
    // ---- B fragments: lane L holds N=L%16, contiguous K half ----
    const int bkb = (lane < 16) ? 0 : 16;
#pragma unroll
    for (int nt = 0; nt < 4; ++nt) {
      const __bf16* row = (const __bf16*)sB[nt * 16 + (lane & 15)];
      v16bf bf;
#pragma unroll
      for (int i = 0; i < 16; ++i) bf[i] = row[bkb + i];
#pragma unroll
      for (int mt = 0; mt < 2; ++mt)
        acc[mt][nt] = __builtin_amdgcn_wmma_f32_16x16x32_bf16(
            false, af[mt], false, bf, (short)0, acc[mt][nt], false, false);
    }
    __syncthreads();
  }

  float g = 0.f;
  if (epi == 2) g = 1.f / (1.f + __expf(-skipptr[0]));

  // C/D layout: VGPR r -> M = r (lanes 0-15) / r+8 (lanes 16-31), N = lane%16
#pragma unroll
  for (int mt = 0; mt < 2; ++mt) {
#pragma unroll
    for (int nt = 0; nt < 4; ++nt) {
      const int gn = n0 + nt * 16 + (lane & 15);
#pragma unroll
      for (int r = 0; r < 8; ++r) {
        const int gm = m0 + wave * 32 + mt * 16 + r + ((lane < 16) ? 0 : 8);
        if (gm < M) {
          float v = acc[mt][nt][r] + bias[gn];
          if (epi == 1)      v = fmaxf(v, 0.f);
          else if (epi == 2) v = g * v + (1.f - g) * hold[(size_t)gm * Nf + gn];
          C[(size_t)gm * Nf + gn] = v;
        }
      }
    }
  }
}

// ---------------------------------------------------------------------------
// Fold relation matrix into projection: Wc[c, h*16+e] = sum_d W[c,h*16+d]*R[h,d,e]
// grid 129 blocks (row 128 == bias), 128 threads (one per output col)
// ---------------------------------------------------------------------------
__global__ void combine_rel(const float* __restrict__ W, const float* __restrict__ b,
                            const float* __restrict__ R,
                            float* __restrict__ Wc, float* __restrict__ bc) {
  int c = blockIdx.x;
  int j = threadIdx.x;
  int h = j >> 4, ep = j & 15;
  const float* src = (c < HIDC) ? (W + (size_t)c * HIDC) : b;
  float acc = 0.f;
#pragma unroll
  for (int d = 0; d < 16; ++d)
    acc += src[h * 16 + d] * R[(h * 16 + d) * 16 + ep];
  if (c < HIDC) Wc[(size_t)c * HIDC + j] = acc;
  else          bc[j] = acc;
}

// ---------------------------------------------------------------------------
__global__ void fill_f32(float* p, float v, size_t n) {
  size_t i = (size_t)blockIdx.x * blockDim.x + threadIdx.x;
  if (i < n) p[i] = v;
}
__global__ void fill_i32(int* p, int v, size_t n) {
  size_t i = (size_t)blockIdx.x * blockDim.x + threadIdx.x;
  if (i < n) p[i] = v;
}

// ---------------------------------------------------------------------------
// Pass 1: raw score per (edge, head) + segment max via ordered-int atomics.
// 8 consecutive threads = one edge -> 512B coalesced gathers of q/kk rows.
// ---------------------------------------------------------------------------
__global__ void edge_score_max(const float* __restrict__ q, const float* __restrict__ kk,
                               const int* __restrict__ ei, const float* __restrict__ prelH,
                               float* __restrict__ alpha, int* __restrict__ maxb, int E) {
  int tid = blockIdx.x * blockDim.x + threadIdx.x;
  if (tid >= E * NHEAD) return;
  int e = tid >> 3, h = tid & 7;
  int src = ei[e], dst = ei[E + e];
  const float4* qp = (const float4*)(q  + (size_t)dst * HIDC + h * 16);
  const float4* kp = (const float4*)(kk + (size_t)src * HIDC + h * 16);
  float s = 0.f;
#pragma unroll
  for (int i = 0; i < 4; ++i) {
    float4 a = qp[i], b = kp[i];
    s += a.x * b.x + a.y * b.y + a.z * b.z + a.w * b.w;
  }
  s *= prelH[h] * 0.25f;                 // scale = 1/sqrt(16)
  alpha[tid] = s;
  int* addr = maxb + (size_t)dst * NHEAD + h;
  if (s >= 0.f) atomicMax(addr, __float_as_int(s));
  else          atomicMin((unsigned int*)addr, __float_as_uint(s));
}

// Pass 2: ex = exp(s - max); segment sum (native global_atomic_add_f32)
__global__ void edge_expsum(const int* __restrict__ ei, const int* __restrict__ maxb,
                            float* __restrict__ alpha, float* __restrict__ sumb, int E) {
  int tid = blockIdx.x * blockDim.x + threadIdx.x;
  if (tid >= E * NHEAD) return;
  int e = tid >> 3, h = tid & 7;
  int dst = ei[E + e];
  float m  = __int_as_float(maxb[(size_t)dst * NHEAD + h]);
  float ex = __expf(alpha[tid] - m);
  alpha[tid] = ex;
  __hip_atomic_fetch_add(sumb + (size_t)dst * NHEAD + h, ex,
                         __ATOMIC_RELAXED, __HIP_MEMORY_SCOPE_AGENT);
}

// Pass 3: wave per edge; scatter-add alpha * vv[src] into out[dst]
__global__ void edge_msg(const int* __restrict__ ei, const float* __restrict__ vv,
                         const float* __restrict__ alpha, const float* __restrict__ sumb,
                         float* __restrict__ out, int E) {
  int gid  = blockIdx.x * blockDim.x + threadIdx.x;
  int e    = gid >> 5;
  int lane = gid & 31;
  if (e >= E) return;
  int src = ei[e], dst = ei[E + e];
  int h = lane >> 2;                                 // cols lane*4..+3 in head h
  float al = alpha[(size_t)e * NHEAD + h] /
             (sumb[(size_t)dst * NHEAD + h] + 1e-16f);
  float4 v = ((const float4*)(vv + (size_t)src * HIDC))[lane];
  float* o = out + (size_t)dst * HIDC + lane * 4;
  __hip_atomic_fetch_add(o + 0, al * v.x, __ATOMIC_RELAXED, __HIP_MEMORY_SCOPE_AGENT);
  __hip_atomic_fetch_add(o + 1, al * v.y, __ATOMIC_RELAXED, __HIP_MEMORY_SCOPE_AGENT);
  __hip_atomic_fetch_add(o + 2, al * v.z, __ATOMIC_RELAXED, __HIP_MEMORY_SCOPE_AGENT);
  __hip_atomic_fetch_add(o + 3, al * v.w, __ATOMIC_RELAXED, __HIP_MEMORY_SCOPE_AGENT);
}

// L2-normalize rows of h (wave per node, wave32 shuffle reduction)
__global__ void l2norm(const float* __restrict__ h, float* __restrict__ out, int N) {
  int gid  = blockIdx.x * blockDim.x + threadIdx.x;
  int n    = gid >> 5;
  int lane = gid & 31;
  if (n >= N) return;
  float4 v = ((const float4*)(h + (size_t)n * HIDC))[lane];
  float ss = v.x * v.x + v.y * v.y + v.z * v.z + v.w * v.w;
#pragma unroll
  for (int o = 16; o > 0; o >>= 1) ss += __shfl_xor(ss, o, 32);
  float inv = 1.f / fmaxf(sqrtf(ss), 1e-12f);
  float4 r = make_float4(v.x * inv, v.y * inv, v.z * inv, v.w * inv);
  ((float4*)(out + (size_t)n * HIDC))[lane] = r;
}

// ---------------------------------------------------------------------------
extern "C" void kernel_launch(void* const* d_in, const int* in_sizes, int n_in,
                              void* d_out, int out_size, void* d_ws, size_t ws_size,
                              hipStream_t stream) {
  const float* x     = (const float*)d_in[0];
  const float* encW  = (const float*)d_in[1];
  const float* encb  = (const float*)d_in[2];
  const float* kW    = (const float*)d_in[3];
  const float* kb    = (const float*)d_in[4];
  const float* qW    = (const float*)d_in[5];
  const float* qb    = (const float*)d_in[6];
  const float* vW    = (const float*)d_in[7];
  const float* vb    = (const float*)d_in[8];
  const float* aW    = (const float*)d_in[9];
  const float* ab    = (const float*)d_in[10];
  const float* skip  = (const float*)d_in[11];
  const float* arel  = (const float*)d_in[12];
  const float* mrel  = (const float*)d_in[13];
  const float* prel  = (const float*)d_in[14];
  const float* decW1 = (const float*)d_in[15];
  const float* decb1 = (const float*)d_in[16];
  const float* decW2 = (const float*)d_in[17];
  const float* decb2 = (const float*)d_in[18];
  const int* ei[4] = {(const int*)d_in[19], (const int*)d_in[20],
                      (const int*)d_in[21], (const int*)d_in[22]};
  const int est[4] = {0, 1, 0, 1};   // source type per edge type (ab,ba,aa,bb)
  const int edt[4] = {1, 0, 0, 1};   // dest type

  const int INC = 256;
  const int N = in_sizes[0] / (2 * INC);
  const int E = in_sizes[19] / 2;
  const size_t NH = (size_t)N * HIDC;

  // workspace carve-out
  float* ws = (float*)d_ws;
  size_t off = 0;
  auto alloc = [&](size_t n) { float* p = ws + off; off += n; return p; };
  float* h[2]    = {alloc(NH), alloc(NH)};
  float* q[2]    = {alloc(NH), alloc(NH)};
  float* kkb     = alloc(NH);
  float* vvb     = alloc(NH);
  float* outb[2] = {alloc(NH), alloc(NH)};
  float* alphab  = alloc((size_t)E * NHEAD);
  float* dtmp    = alloc(NH);
  float* cW1 = alloc(HIDC * HIDC); float* cb1 = alloc(HIDC);
  float* cW2 = alloc(HIDC * HIDC); float* cb2 = alloc(HIDC);
  int*   maxb = (int*)alloc((size_t)N * NHEAD);
  float* sumb = alloc((size_t)N * NHEAD);

  auto gemm = [&](const float* A, const float* Wm, const float* bias, float* C,
                  int M, int K, int Nf, int a_act, int epi,
                  const float* hold, const float* sp) {
    dim3 grid((M + BM - 1) / BM, Nf / BN);
    gemm_bf16_wmma<<<grid, 128, 0, stream>>>(A, Wm, bias, C, M, K, Nf,
                                             a_act, epi, hold, sp);
  };

  // encoder: h[t] = relu(x[t] @ encW[t] + encb[t])
  for (int t = 0; t < 2; ++t)
    gemm(x + (size_t)t * N * INC, encW + (size_t)t * INC * HIDC,
         encb + t * HIDC, h[t], N, INC, HIDC, 0, 1, nullptr, nullptr);

  const size_t nh8 = (size_t)N * NHEAD;
  const size_t eh  = (size_t)E * NHEAD;
  const size_t ew  = (size_t)E * 32;

  for (int l = 0; l < 2; ++l) {
    for (int t = 0; t < 2; ++t)
      gemm(h[t], qW + (size_t)(l * 2 + t) * HIDC * HIDC,
           qb + (l * 2 + t) * HIDC, q[t], N, HIDC, HIDC, 0, 0, nullptr, nullptr);
    for (int t = 0; t < 2; ++t)
      fill_f32<<<(unsigned)((NH + 255) / 256), 256, 0, stream>>>(outb[t], 0.f, NH);

    for (int e4 = 0; e4 < 4; ++e4) {
      int st = est[e4], dt = edt[e4];
      const float* Rk = arel + (size_t)(l * 4 + e4) * NHEAD * 256;
      const float* Rv = mrel + (size_t)(l * 4 + e4) * NHEAD * 256;
      combine_rel<<<129, 128, 0, stream>>>(kW + (size_t)(l * 2 + st) * HIDC * HIDC,
                                           kb + (l * 2 + st) * HIDC, Rk, cW1, cb1);
      combine_rel<<<129, 128, 0, stream>>>(vW + (size_t)(l * 2 + st) * HIDC * HIDC,
                                           vb + (l * 2 + st) * HIDC, Rv, cW2, cb2);
      gemm(h[st], cW1, cb1, kkb, N, HIDC, HIDC, 0, 0, nullptr, nullptr);
      gemm(h[st], cW2, cb2, vvb, N, HIDC, HIDC, 0, 0, nullptr, nullptr);

      fill_i32<<<(unsigned)((nh8 + 255) / 256), 256, 0, stream>>>(maxb, (int)0xFF800000u, nh8);
      fill_f32<<<(unsigned)((nh8 + 255) / 256), 256, 0, stream>>>(sumb, 0.f, nh8);

      edge_score_max<<<(unsigned)((eh + 255) / 256), 256, 0, stream>>>(
          q[dt], kkb, ei[e4], prel + (l * 4 + e4) * NHEAD, alphab, maxb, E);
      edge_expsum<<<(unsigned)((eh + 255) / 256), 256, 0, stream>>>(
          ei[e4], maxb, alphab, sumb, E);
      edge_msg<<<(unsigned)((ew + 255) / 256), 256, 0, stream>>>(
          ei[e4], vvb, alphab, sumb, outb[dt], E);
    }

    // h[t] = g * (gelu(out[t]) @ aW + ab) + (1-g) * h[t]
    for (int t = 0; t < 2; ++t)
      gemm(outb[t], aW + (size_t)(l * 2 + t) * HIDC * HIDC,
           ab + (l * 2 + t) * HIDC, h[t], N, HIDC, HIDC, 1, 2,
           h[t], skip + l * 2 + t);
  }

  float* emb = (float*)d_out;
  float* rec = (float*)d_out + NH;
  l2norm<<<(unsigned)(((size_t)N * 32 + 255) / 256), 256, 0, stream>>>(h[0], emb, N);
  gemm(h[0], decW1, decb1, dtmp, N, HIDC, HIDC, 0, 1, nullptr, nullptr);
  gemm(dtmp, decW2, decb2, rec, N, HIDC, INC, 0, 0, nullptr, nullptr);

  (void)n_in; (void)out_size; (void)ws_size;
}
